// SparseAttention_40948218200612
// MI455X (gfx1250) — matrix operand order repaired and checked by reading, not compiled, not run on hardware
//
#include <hip/hip_runtime.h>

// ---------------------------------------------------------------------------
// MI455X (gfx1250) memory-augmented local attention, bf16 WMMA pipeline.
//   0) one-time: W_qkv / W_proj -> bf16 transposed [N][K]; memory -> bf16
//   1) qkv = x @ W_qkv + b   (bf16 WMMA, fp32 accum) -> q/k/v bf16 workspaces
//   2) attention per (b, head, 64-row tile); fixed 256-key resident tile;
//      epilogue fuses 1/rowsum and the per-column gate
//   3) out = att @ W_proj + b_proj, * loss_mask  (bf16 WMMA, pure-copy loaders)
// ---------------------------------------------------------------------------

typedef __bf16 v16bf __attribute__((ext_vector_type(16)));
typedef float  v8f   __attribute__((ext_vector_type(8)));

union FragU { v16bf v; uint4 q[2]; };
union HU    { uint4 q; unsigned short h[8]; unsigned int u[4]; };
union VU    { uint4 q[8]; unsigned short h[64]; };

#define BATCH 512
#define SEQT  256
#define CDIM  384
#define MMEM  128
#define NHEAD 6
#define DHEAD 64
#define NROWS (BATCH * SEQT)

// round-to-nearest (ties away) fp32 -> bf16: 2 VALU per element
__device__ __forceinline__ unsigned short f2bf(float f) {
  return (unsigned short)((__float_as_uint(f) + 0x8000u) >> 16);
}

// pack two fp32 into one dword of bf16 (lo=a, hi=b): ~4 VALU per pair
__device__ __forceinline__ unsigned int pack2bf(float a, float b) {
  unsigned int ua = __float_as_uint(a) + 0x8000u;
  unsigned int ub = __float_as_uint(b) + 0x8000u;
  return (ua >> 16) | (ub & 0xffff0000u);
}

// A-matrix 16x32 bf16 fragment: lane m=lane%16, half=lane>>4.
// K chunks: [half*8, half*8+8) and [16+half*8, ...): two 16B contiguous loads.
__device__ __forceinline__ v16bf load_frag_a(const unsigned short* base, int stride, int lane) {
  int m = lane & 15, half = lane >> 4;
  FragU f;
  const unsigned short* p = base + m * stride + half * 8;
  f.q[0] = *(const uint4*)(p);
  f.q[1] = *(const uint4*)(p + 16);
  return f.v;
}

// B-matrix 32x16 bf16 fragment from n-major [n][k] storage: lane n=lane%16,
// K chunks: [half*8, half*8+8) and [16+half*8, ...): two 16B contiguous loads.
__device__ __forceinline__ v16bf load_frag_b(const unsigned short* base, int stride, int lane) {
  int n = lane & 15, half = lane >> 4;
  FragU f;
  const unsigned short* p = base + n * stride + half * 8;
  f.q[0] = *(const uint4*)(p);
  f.q[1] = *(const uint4*)(p + 16);
  return f.v;
}

__device__ __forceinline__ v8f wmma_bf16(v16bf a, v16bf b, v8f c) {
  return __builtin_amdgcn_wmma_f32_16x16x32_bf16(false, a, false, b, (short)0, c, false, false);
}

// ---------------------------------------------------------------------------
// One-time prep: memory -> bf16 (packed), and W -> bf16 transposed [N][K].
// ---------------------------------------------------------------------------
__global__ __launch_bounds__(256) void mem_to_bf16(const float* __restrict__ src,
                                                   unsigned short* __restrict__ dst) {
  int i = (blockIdx.x * 256 + threadIdx.x) * 2;
  if (i < MMEM * CDIM)
    *(unsigned int*)&dst[i] = pack2bf(src[i], src[i + 1]);
}

// Wt[n*K + k] = bf16(W[k*N + n]); two adjacent k per thread (packed store).
__global__ __launch_bounds__(256) void transpose_w_bf16(const float* __restrict__ W,
                                                        unsigned short* __restrict__ Wt,
                                                        int K, int N) {
  int idx = (blockIdx.x * 256 + threadIdx.x) * 2;
  if (idx >= K * N) return;
  int n = idx / K, k = idx % K;
  *(unsigned int*)&Wt[idx] =
      pack2bf(W[(size_t)k * N + n], W[(size_t)(k + 1) * N + n]);
}

// ---------------------------------------------------------------------------
// Kernel 2: QKV GEMM.  x[131072,384] (fp32) @ W_qkv (pre-transposed bf16).
// 128x128 block tile, BK=64, 8 waves (wm 0..3, wn 0..1), wave = 32x64.
// ---------------------------------------------------------------------------
__global__ __launch_bounds__(256) void qkv_gemm(const float* __restrict__ x,
                                                const unsigned short* __restrict__ Wt,
                                                const float* __restrict__ bias,
                                                unsigned short* __restrict__ qws,
                                                unsigned short* __restrict__ kws,
                                                unsigned short* __restrict__ vws) {
  const int K = CDIM, BK = 64;
  __shared__ __align__(16) unsigned short As[128 * 64];   // [row][k]
  __shared__ __align__(16) unsigned short Bt[128 * 64];   // [n][k]

  int tid = threadIdx.x;
  int wave = tid >> 5, lane = tid & 31;
  int wm = wave >> 1, wn = wave & 1;
  int rowBase = blockIdx.x * 128;
  int colBase = blockIdx.y * 128;

  v8f acc[2][4];
#pragma unroll
  for (int mt = 0; mt < 2; ++mt)
#pragma unroll
    for (int nt = 0; nt < 4; ++nt)
      acc[mt][nt] = (v8f){0.f, 0.f, 0.f, 0.f, 0.f, 0.f, 0.f, 0.f};

  int arow = tid >> 1, acs = (tid & 1) * 32;   // A: 32 floats/thread
  int brow = tid >> 1, bks = (tid & 1) * 32;   // B: 32 bf16/thread (pure copy)

#pragma unroll
  for (int k0 = 0; k0 < K; k0 += BK) {
    {  // A tile: fp32 -> bf16 (packed), 4x uint4 LDS stores
      const float* src = x + (size_t)(rowBase + arow) * K + k0 + acs;
      uint4* dst = (uint4*)&As[arow * BK + acs];
#pragma unroll
      for (int c = 0; c < 4; ++c) {
        HU o;
#pragma unroll
        for (int j = 0; j < 4; ++j)
          o.u[j] = pack2bf(src[c * 8 + 2 * j], src[c * 8 + 2 * j + 1]);
        dst[c] = o.q;
      }
      if (k0 + BK < K) __builtin_prefetch(src + BK, 0, 3);
    }
    {  // B tile: straight b128 copy of pre-transposed bf16 weights
      const unsigned short* src = Wt + (size_t)(colBase + brow) * K + k0 + bks;
      uint4* dst = (uint4*)&Bt[brow * BK + bks];
#pragma unroll
      for (int c = 0; c < 4; ++c) dst[c] = ((const uint4*)src)[c];
    }
    __syncthreads();

    v16bf aF[2][2], bF[2][4];
#pragma unroll
    for (int ks = 0; ks < 2; ++ks) {
#pragma unroll
      for (int mt = 0; mt < 2; ++mt)
        aF[ks][mt] = load_frag_a(As + (wm * 32 + mt * 16) * BK + ks * 32, BK, lane);
#pragma unroll
      for (int nt = 0; nt < 4; ++nt)
        bF[ks][nt] = load_frag_b(Bt + (wn * 64 + nt * 16) * BK + ks * 32, BK, lane);
    }
#pragma unroll
    for (int ks = 0; ks < 2; ++ks)
#pragma unroll
      for (int nt = 0; nt < 4; ++nt)
#pragma unroll
        for (int mt = 0; mt < 2; ++mt)
          acc[mt][nt] = wmma_bf16(aF[ks][mt], bF[ks][nt], acc[mt][nt]);
    __syncthreads();
  }

  // branchless epilogue: dst buffer uniform for the whole block
  int third = colBase / CDIM;                              // 0,1,2 (uniform)
  unsigned short* dst = (third == 0) ? qws : ((third == 1) ? kws : vws);
  int cb = colBase - third * CDIM;

  int nl = lane & 15, half = lane >> 4;
  float bv4[4];
#pragma unroll
  for (int nt = 0; nt < 4; ++nt)
    bv4[nt] = bias[colBase + wn * 64 + nt * 16 + nl];

#pragma unroll
  for (int mt = 0; mt < 2; ++mt)
#pragma unroll
    for (int nt = 0; nt < 4; ++nt)
#pragma unroll
      for (int r = 0; r < 8; ++r) {
        int gm = rowBase + wm * 32 + mt * 16 + half * 8 + r;
        int cc = cb + wn * 64 + nt * 16 + nl;
        dst[(size_t)gm * CDIM + cc] = f2bf(acc[mt][nt][r] + bv4[nt]);
      }
}

// ---------------------------------------------------------------------------
// Kernel 3: attention.  One block per (b, head, 64-row tile). 4 waves; each
// wave owns 16 rows. 256 resident key columns: [0,128)=memory (always valid),
// [128,256)=self keys starting at selfBase=max(0,i0-64), masked causal+window.
// Epilogue applies 1/rowsum AND the per-column gate (fp32, pre-rounding).
// ---------------------------------------------------------------------------
__global__ __launch_bounds__(128) void attn_kernel(const unsigned short* __restrict__ qws,
                                                   const unsigned short* __restrict__ kws,
                                                   const unsigned short* __restrict__ vws,
                                                   const unsigned short* __restrict__ membf,
                                                   const float* __restrict__ gate,
                                                   unsigned short* __restrict__ aws) {
  __shared__ __align__(16) unsigned short Ks[256 * 64];   // later aliased as P
  __shared__ __align__(16) unsigned short Vt[64 * 256];   // [d][col]

  int idx = blockIdx.x;
  int tq = idx & 3;
  int hd = (idx >> 2) % NHEAD;
  int b  = idx / (4 * NHEAD);
  int i0 = tq * 64;
  int selfBase = (i0 >= 64) ? (i0 - 64) : 0;

  int tid = threadIdx.x, wave = tid >> 5, lane = tid & 31;
  int nl = lane & 15, half = lane >> 4;

  // ---- cooperative K / V tile load: 2 adjacent columns per thread ----
  auto colptr = [&](int col, const unsigned short*& kp, const unsigned short*& vp) {
    if (col < MMEM) {
      kp = membf + (size_t)col * CDIM + hd * DHEAD;
      vp = kp;
    } else {
      size_t off = ((size_t)(b * SEQT + selfBase + col - MMEM)) * CDIM + hd * DHEAD;
      kp = kws + off;
      vp = vws + off;
    }
  };
  {
    int c0 = tid * 2;
    const unsigned short *kp0, *vp0, *kp1, *vp1;
    colptr(c0, kp0, vp0);
    colptr(c0 + 1, kp1, vp1);

    uint4* kd0 = (uint4*)&Ks[(size_t)c0 * DHEAD];
    uint4* kd1 = (uint4*)&Ks[(size_t)(c0 + 1) * DHEAD];
#pragma unroll
    for (int q = 0; q < 8; ++q) kd0[q] = ((const uint4*)kp0)[q];
#pragma unroll
    for (int q = 0; q < 8; ++q) kd1[q] = ((const uint4*)kp1)[q];

    // V transposed: pack two adjacent columns into one dword per d
    VU v0, v1;
#pragma unroll
    for (int q = 0; q < 8; ++q) v0.q[q] = ((const uint4*)vp0)[q];
#pragma unroll
    for (int q = 0; q < 8; ++q) v1.q[q] = ((const uint4*)vp1)[q];
    unsigned int* Vt32 = (unsigned int*)Vt;
#pragma unroll
    for (int dd = 0; dd < DHEAD; ++dd)
      Vt32[dd * 128 + tid] = (unsigned int)v0.h[dd] | ((unsigned int)v1.h[dd] << 16);
  }

  // ---- Q fragments straight from global (layout already matches A-frag) ----
  const unsigned short* qbase =
      qws + ((size_t)(b * SEQT + i0 + wave * 16)) * CDIM + hd * DHEAD;
  v16bf aQ[2];
  aQ[0] = load_frag_a(qbase, CDIM, lane);
  aQ[1] = load_frag_a(qbase + 32, CDIM, lane);
  __syncthreads();

  // ---- S = Q K^T : 16 n-tiles x 2 k-steps ----
  v8f s[16];
#pragma unroll
  for (int nt = 0; nt < 16; ++nt)
    s[nt] = (v8f){0.f, 0.f, 0.f, 0.f, 0.f, 0.f, 0.f, 0.f};
#pragma unroll
  for (int nt = 0; nt < 16; ++nt) {
    v16bf bK0 = load_frag_b(Ks + nt * 16 * DHEAD, DHEAD, lane);
    v16bf bK1 = load_frag_b(Ks + nt * 16 * DHEAD + 32, DHEAD, lane);
    s[nt] = wmma_bf16(aQ[0], bK0, s[nt]);
    s[nt] = wmma_bf16(aQ[1], bK1, s[nt]);
  }

  // ---- mask + one-pass softmax (memory cols keep rowmax finite) ----
  const float scale = 0.125f;   // 1/sqrt(64)
  float rmax[8], rsum[8];
#pragma unroll
  for (int r = 0; r < 8; ++r) rmax[r] = -INFINITY;
#pragma unroll
  for (int nt = 0; nt < 16; ++nt)
#pragma unroll
    for (int r = 0; r < 8; ++r) {
      int i = i0 + wave * 16 + half * 8 + r;
      int c = nt * 16 + nl;
      bool valid = (c < MMEM);
      if (!valid) {
        int j = selfBase + c - MMEM;
        valid = (j <= i) && (i - j <= 64);
      }
      float v = valid ? s[nt][r] * scale : -INFINITY;
      s[nt][r] = v;
      rmax[r] = fmaxf(rmax[r], v);
    }
#pragma unroll
  for (int r = 0; r < 8; ++r)
#pragma unroll
    for (int m = 1; m < 16; m <<= 1)
      rmax[r] = fmaxf(rmax[r], __shfl_xor(rmax[r], m, 16));
#pragma unroll
  for (int r = 0; r < 8; ++r) rsum[r] = 0.f;
#pragma unroll
  for (int nt = 0; nt < 16; ++nt)
#pragma unroll
    for (int r = 0; r < 8; ++r) {
      float p = __expf(s[nt][r] - rmax[r]);
      s[nt][r] = p;
      rsum[r] += p;
    }
#pragma unroll
  for (int r = 0; r < 8; ++r)
#pragma unroll
    for (int m = 1; m < 16; m <<= 1)
      rsum[r] += __shfl_xor(rsum[r], m, 16);

  // ---- stage P to LDS (alias over Ks) in A-frag friendly row-major ----
  __syncthreads();                    // all waves done reading Ks
  unsigned short* Ps = Ks;
#pragma unroll
  for (int nt = 0; nt < 16; ++nt)
#pragma unroll
    for (int r = 0; r < 8; ++r)
      Ps[(wave * 16 + half * 8 + r) * 256 + nt * 16 + nl] = f2bf(s[nt][r]);
  __syncthreads();

  // ---- O = P V : 4 d-tiles x 8 k-steps ----
  v8f o[4];
#pragma unroll
  for (int dn = 0; dn < 4; ++dn)
    o[dn] = (v8f){0.f, 0.f, 0.f, 0.f, 0.f, 0.f, 0.f, 0.f};
#pragma unroll
  for (int kt = 0; kt < 8; ++kt) {
    v16bf aP = load_frag_a(Ps + wave * 16 * 256 + kt * 32, 256, lane);
#pragma unroll
    for (int dn = 0; dn < 4; ++dn) {
      v16bf bV = load_frag_b(Vt + dn * 16 * 256 + kt * 32, 256, lane);
      o[dn] = wmma_bf16(aP, bV, o[dn]);
    }
  }

  // ---- normalize, apply gate (per column), store bf16 ----
  float g4[4];
#pragma unroll
  for (int dn = 0; dn < 4; ++dn)
    g4[dn] = gate[hd * DHEAD + dn * 16 + nl];

#pragma unroll
  for (int r = 0; r < 8; ++r) {
    float rinv = 1.0f / rsum[r];
    size_t row = (size_t)(b * SEQT + i0 + wave * 16 + half * 8 + r);
#pragma unroll
    for (int dn = 0; dn < 4; ++dn)
      aws[row * CDIM + hd * DHEAD + dn * 16 + nl] = f2bf(o[dn][r] * rinv * g4[dn]);
  }
}

// ---------------------------------------------------------------------------
// Kernel 4: output projection. att (gate pre-applied) @ W_proj (pre-transposed
// bf16) + b_proj, times loss_mask, fp32 out. Pure-copy loaders, BK=64.
// ---------------------------------------------------------------------------
__global__ __launch_bounds__(256) void proj_gemm(const unsigned short* __restrict__ att,
                                                 const unsigned short* __restrict__ Wt,
                                                 const float* __restrict__ bias,
                                                 const float* __restrict__ lmask,
                                                 float* __restrict__ out) {
  const int K = CDIM, BK = 64;
  __shared__ __align__(16) unsigned short As[128 * 64];
  __shared__ __align__(16) unsigned short Bt[128 * 64];

  int tid = threadIdx.x;
  int wave = tid >> 5, lane = tid & 31;
  int wm = wave >> 1, wn = wave & 1;
  int rowBase = blockIdx.x * 128;
  int colBase = blockIdx.y * 128;

  v8f acc[2][4];
#pragma unroll
  for (int mt = 0; mt < 2; ++mt)
#pragma unroll
    for (int nt = 0; nt < 4; ++nt)
      acc[mt][nt] = (v8f){0.f, 0.f, 0.f, 0.f, 0.f, 0.f, 0.f, 0.f};

  int arow = tid >> 1, acs = (tid & 1) * 32;
  int brow = tid >> 1, bks = (tid & 1) * 32;

#pragma unroll
  for (int k0 = 0; k0 < K; k0 += BK) {
    {  // A tile: straight b128 copy (gate already applied upstream)
      const unsigned short* src = att + (size_t)(rowBase + arow) * K + k0 + acs;
      uint4* dst = (uint4*)&As[arow * BK + acs];
#pragma unroll
      for (int c = 0; c < 4; ++c) dst[c] = ((const uint4*)src)[c];
      if (k0 + BK < K) __builtin_prefetch(src + BK, 0, 3);
    }
    {  // B tile: straight b128 copy of pre-transposed bf16 weights
      const unsigned short* src = Wt + (size_t)(colBase + brow) * K + k0 + bks;
      uint4* dst = (uint4*)&Bt[brow * BK + bks];
#pragma unroll
      for (int c = 0; c < 4; ++c) dst[c] = ((const uint4*)src)[c];
    }
    __syncthreads();

    v16bf aF[2][2], bF[2][4];
#pragma unroll
    for (int ks = 0; ks < 2; ++ks) {
#pragma unroll
      for (int mt = 0; mt < 2; ++mt)
        aF[ks][mt] = load_frag_a(As + (wm * 32 + mt * 16) * BK + ks * 32, BK, lane);
#pragma unroll
      for (int nt = 0; nt < 4; ++nt)
        bF[ks][nt] = load_frag_b(Bt + (wn * 64 + nt * 16) * BK + ks * 32, BK, lane);
    }
#pragma unroll
    for (int ks = 0; ks < 2; ++ks)
#pragma unroll
      for (int nt = 0; nt < 4; ++nt)
#pragma unroll
        for (int mt = 0; mt < 2; ++mt)
          acc[mt][nt] = wmma_bf16(aF[ks][mt], bF[ks][nt], acc[mt][nt]);
    __syncthreads();
  }

  int nl = lane & 15, half = lane >> 4;
  float bv4[4];
#pragma unroll
  for (int nt = 0; nt < 4; ++nt)
    bv4[nt] = bias[colBase + wn * 64 + nt * 16 + nl];
  float lm[2][8];
#pragma unroll
  for (int mt = 0; mt < 2; ++mt)
#pragma unroll
    for (int r = 0; r < 8; ++r)
      lm[mt][r] = lmask[rowBase + wm * 32 + mt * 16 + half * 8 + r];

#pragma unroll
  for (int mt = 0; mt < 2; ++mt)
#pragma unroll
    for (int nt = 0; nt < 4; ++nt)
#pragma unroll
      for (int r = 0; r < 8; ++r) {
        int gm = rowBase + wm * 32 + mt * 16 + half * 8 + r;
        int gn = colBase + wn * 64 + nt * 16 + nl;
        out[(size_t)gm * CDIM + gn] = (acc[mt][nt][r] + bv4[nt]) * lm[mt][r];
      }
}

// ---------------------------------------------------------------------------
extern "C" void kernel_launch(void* const* d_in, const int* in_sizes, int n_in,
                              void* d_out, int out_size, void* d_ws, size_t ws_size,
                              hipStream_t stream) {
  (void)in_sizes; (void)n_in; (void)out_size; (void)ws_size;

  const float* x         = (const float*)d_in[0];
  const float* memory    = (const float*)d_in[1];
  const float* loss_mask = (const float*)d_in[2];
  const float* W_qkv     = (const float*)d_in[3];
  const float* b_qkv     = (const float*)d_in[4];
  const float* W_proj    = (const float*)d_in[5];
  const float* b_proj    = (const float*)d_in[6];
  const float* gate      = (const float*)d_in[7];
  float* out = (float*)d_out;

  // Workspace layout (bf16 = unsigned short), ~404 MB total:
  const size_t plane = (size_t)NROWS * CDIM;          // 131072*384
  unsigned short* q_ws   = (unsigned short*)d_ws;
  unsigned short* k_ws   = q_ws + plane;
  unsigned short* v_ws   = k_ws + plane;
  unsigned short* a_ws   = v_ws + plane;
  unsigned short* membf  = a_ws + plane;              // 128*384
  unsigned short* wqkv_t = membf + (size_t)MMEM * CDIM;         // [1152][384]
  unsigned short* wproj_t = wqkv_t + (size_t)(3 * CDIM) * CDIM; // [384][384]

  mem_to_bf16<<<(MMEM * CDIM / 2 + 255) / 256, 256, 0, stream>>>(memory, membf);
  transpose_w_bf16<<<(CDIM * 3 * CDIM / 2 + 255) / 256, 256, 0, stream>>>(
      W_qkv, wqkv_t, CDIM, 3 * CDIM);
  transpose_w_bf16<<<(CDIM * CDIM / 2 + 255) / 256, 256, 0, stream>>>(
      W_proj, wproj_t, CDIM, CDIM);

  qkv_gemm<<<dim3(NROWS / 128, (3 * CDIM) / 128), 256, 0, stream>>>(
      x, wqkv_t, b_qkv, q_ws, k_ws, v_ws);
  attn_kernel<<<BATCH * NHEAD * (SEQT / 64), 128, 0, stream>>>(
      q_ws, k_ws, v_ws, membf, gate, a_ws);
  proj_gemm<<<dim3(NROWS / 128, CDIM / 128), 256, 0, stream>>>(
      a_ws, wproj_t, b_proj, loss_mask, out);
}
